// EquiformerV2_MatPES_71923522339385
// MI455X (gfx1250) — matrix operand order repaired and checked by reading, not compiled, not run on hardware
//
#include <hip/hip_runtime.h>
#include <hip/hip_bf16.h>

typedef __attribute__((ext_vector_type(16))) _Float16 v16h;
typedef __attribute__((ext_vector_type(8)))  _Float16 v8h;
typedef __attribute__((ext_vector_type(8)))  float    v8f;
typedef __attribute__((ext_vector_type(4)))  int      v4i;

#define NATOMS 10000
#define NEDGES 120000
#define NGRAPH 64
#define CDIM   128
#define NCH    9
#define NLAY   6
#define FDIM   512
#define DRBF   600
#define DPAD   608

// ---- LDS staging geometry for the generic GEMM ----
#define KCHUNK 256
#define LDSP   (KCHUNK + 8)            // pitch in halves; 528B rows -> 4-bank rotation
#define GEMM_LDS_BYTES (LDSP * 64 * 2) // 33792 B
#define MSGP   (128 + 8)
#define MSG_LDS_BYTES (2 * MSGP * 64 * 2) // 34816 B

// ---- async global->LDS (CDNA5) with synchronous fallback ----
#if __has_builtin(__builtin_amdgcn_global_load_async_to_lds_b128) && \
    __has_builtin(__builtin_amdgcn_s_wait_asynccnt)
#define ASYNC_LDS 1
typedef __attribute__((address_space(1))) v4i* as1_v4i;
typedef __attribute__((address_space(3))) v4i* as3_v4i;
#define TO_AS1(p) ((as1_v4i)(unsigned long long)(const void*)(p))
#define TO_AS3(p) ((as3_v4i)(unsigned)(unsigned long long)(void*)(p))
#else
#define ASYNC_LDS 0
#endif

__device__ __forceinline__ void lds_copy16(const _Float16* gsrc, _Float16* ldst) {
#if ASYNC_LDS
  __builtin_amdgcn_global_load_async_to_lds_b128(TO_AS1(gsrc), TO_AS3(ldst), 0, 0);
#else
  *(v8h*)ldst = *(const v8h*)gsrc;
#endif
}
__device__ __forceinline__ void lds_copy_wait() {
#if ASYNC_LDS
  __builtin_amdgcn_s_wait_asynccnt(0);
#endif
}

// ---------------------------------------------------------------------------
// WMMA fragment loaders (layouts per CDNA5 ISA 7.12.2, wave32)
// A 16x32 f16: lane<16 -> row=lane, K={0..7,16..23}; lane>=16 -> row, K={8..15,24..31}
// B 32x16 f16: lane<16 -> col=lane, K={0..7,16..23}; lane>=16 -> K={8..15,24..31}
// C/D 16x16 f32: vgpr j, lane l -> row = j + 8*(l>=16), col = l&15
// ---------------------------------------------------------------------------
__device__ __forceinline__ v16h frag_a16(const _Float16* __restrict__ row, int koff) {
  v8h lo = *(const v8h*)(row + koff);
  v8h hi = *(const v8h*)(row + koff + 16);
  v16h r;
#pragma unroll
  for (int i = 0; i < 8; ++i) { r[i] = lo[i]; r[i + 8] = hi[i]; }
  return r;
}
__device__ __forceinline__ v16h frag_b16(const _Float16* row, int koff) {
  v8h lo = *(const v8h*)(row + koff);
  v8h hi = *(const v8h*)(row + koff + 16);
  v16h r;
#pragma unroll
  for (int i = 0; i < 8; ++i) { r[i] = lo[i]; r[i + 8] = hi[i]; }
  return r;
}
__device__ __forceinline__ v16h frag_a32(const float* __restrict__ row, int koff) {
  v16h r;
#pragma unroll
  for (int i = 0; i < 8; ++i) {
    r[i]     = (_Float16)row[koff + i];
    r[i + 8] = (_Float16)row[koff + 16 + i];
  }
  return r;
}

__device__ __forceinline__ float siluf(float v) { return v / (1.0f + __expf(-v)); }

// ---------------------------------------------------------------------------
// Generic WMMA GEMM:  C(MxN) = A(MxK) * B(KxN),  B stored transposed (N x K) f16.
// B panel (64 cols x Kchunk) staged in LDS per block via async global->LDS.
// wave tile 16x64, block = 8 waves (128 rows), grid = (ceil(M/128), N/64)
// EPI: 0 store f16 | 1 bias+silu f16 | 2 bias f16 | 3 silu f16
//      4 add into f32 | 5 attention-value scatter (x alpha, atomicAdd into agg)
// ---------------------------------------------------------------------------
template<bool AF32, int EPI>
__global__ __launch_bounds__(256) void gemm_k(
    const void* __restrict__ Av, long rowStride,
    const _Float16* __restrict__ Bt, int K, int M,
    const float* __restrict__ bias,
    const float* __restrict__ alphaBuf, const int* __restrict__ dstIdx,
    float* __restrict__ aggOut,
    void* __restrict__ Cout, long ldc,
    float* __restrict__ addOut)
{
  extern __shared__ _Float16 Blds[];
  const int tid  = threadIdx.x;
  const int wave = tid >> 5;
  const int lane = tid & 31;
  const int row0 = (blockIdx.x * 8 + wave) * 16;
  const bool rowok = (row0 < M);
  const int col0 = blockIdx.y * 64;
  const int ksA  = (lane < 16) ? 0 : 8;
  const int ksB  = (lane < 16) ? 0 : 8;

  int ar = row0 + (lane & 15);
  if (ar >= M) ar = M - 1;
  const _Float16* arow  = nullptr;
  const float*    arowf = nullptr;
  if constexpr (AF32) arowf = (const float*)Av + (long)ar * rowStride;
  else                arow  = (const _Float16*)Av + (long)ar * rowStride;

  const _Float16* lb0 = Blds + (lane & 15) * LDSP;
  v8f acc[4] = {};

  for (int kbase = 0; kbase < K; kbase += KCHUNK) {
    const int kc   = (K - kbase < KCHUNK) ? (K - kbase) : KCHUNK;
    const int vper = kc >> 3;  // v8h chunks per column
    __syncthreads();           // previous chunk fully consumed
    for (int c = tid; c < 64 * vper; c += 256) {
      const int i   = c / vper;
      const int off = (c - i * vper) * 8;
      lds_copy16(Bt + (long)(col0 + i) * K + kbase + off, Blds + i * LDSP + off);
    }
    lds_copy_wait();
    __syncthreads();

    for (int k = 0; k < kc; k += 32) {
      if (kbase + k + 32 < K) {
        if constexpr (AF32) __builtin_prefetch(arowf + kbase + k + 32, 0, 0);
        else                __builtin_prefetch(arow + kbase + k + 32, 0, 0);
      }
      v16h a;
      if constexpr (AF32) a = frag_a32(arowf, kbase + k + ksA);
      else                a = frag_a16(arow,  kbase + k + ksA);
      v16h fb0 = frag_b16(lb0,             k + ksB);
      v16h fb1 = frag_b16(lb0 + 16 * LDSP, k + ksB);
      v16h fb2 = frag_b16(lb0 + 32 * LDSP, k + ksB);
      v16h fb3 = frag_b16(lb0 + 48 * LDSP, k + ksB);
      acc[0] = __builtin_amdgcn_wmma_f32_16x16x32_f16(false, a, false, fb0, (short)0, acc[0], false, false);
      acc[1] = __builtin_amdgcn_wmma_f32_16x16x32_f16(false, a, false, fb1, (short)0, acc[1], false, false);
      acc[2] = __builtin_amdgcn_wmma_f32_16x16x32_f16(false, a, false, fb2, (short)0, acc[2], false, false);
      acc[3] = __builtin_amdgcn_wmma_f32_16x16x32_f16(false, a, false, fb3, (short)0, acc[3], false, false);
    }
  }

  if (!rowok) return;
  const int subr = (lane >> 4) << 3;
  const int cloc = lane & 15;
#pragma unroll
  for (int t = 0; t < 4; ++t) {
#pragma unroll
    for (int j = 0; j < 8; ++j) {
      const long row = row0 + subr + j;
      const int  col = col0 + t * 16 + cloc;
      float v = acc[t][j];
      if constexpr (EPI == 1 || EPI == 2) v += bias[col];
      if constexpr (EPI == 1 || EPI == 3) v = siluf(v);
      if constexpr (EPI <= 3) {
        ((_Float16*)Cout)[row * ldc + col] = (_Float16)v;
      } else if constexpr (EPI == 4) {
        addOut[row * ldc + col] += v;
      } else if constexpr (EPI == 5) {
        const int e = (int)(row / 9);
        const int n = (int)(row - 9L * e);
        const float al = alphaBuf[e * 8 + (col >> 4)];
        atomicAdd(&aggOut[((long)dstIdx[e] * 9 + n) * 128 + col], v * al);
      }
    }
  }
}

// ---------------------------------------------------------------------------
// Message GEMM: m[e,n,:] = (y[src[e],n,:]@Wsrc + y[dst[e],n,:]@Wdst) * rad[e,:]
// M = E*9, K = 128, N = 128. Both weight panels staged in LDS.
// ---------------------------------------------------------------------------
__global__ __launch_bounds__(256) void gemm_msg_k(
    const _Float16* __restrict__ Y,
    const int* __restrict__ srcIdx, const int* __restrict__ dstIdx,
    const _Float16* __restrict__ BsT, const _Float16* __restrict__ BdT,
    const _Float16* __restrict__ rad,
    _Float16* __restrict__ Mout, int M)
{
  extern __shared__ _Float16 Wlds[];  // [2][64][MSGP]
  const int tid  = threadIdx.x;
  const int wave = tid >> 5;
  const int lane = tid & 31;
  const int row0 = (blockIdx.x * 8 + wave) * 16;
  const bool rowok = (row0 < M);
  const int col0 = blockIdx.y * 64;
  const int ksA  = (lane < 16) ? 0 : 8;
  const int ksB  = (lane < 16) ? 0 : 8;

  // stage both weight panels (64 cols x 128 halves each)
  for (int c = tid; c < 2 * 64 * 16; c += 256) {
    const int pan = c >> 10;            // 0 = src, 1 = dst (1024 chunks each)
    const int i   = (c & 1023) >> 4;    // column
    const int off = (c & 15) * 8;
    const _Float16* src = (pan ? BdT : BsT) + (long)(col0 + i) * 128 + off;
    lds_copy16(src, Wlds + (pan * 64 + i) * MSGP + off);
  }
  lds_copy_wait();
  __syncthreads();

  int r = row0 + (lane & 15);
  if (r >= M) r = M - 1;
  const int e0 = r / 9;
  const int n0 = r - 9 * e0;
  const _Float16* aS = Y + ((long)srcIdx[e0] * 9 + n0) * 128;
  const _Float16* aD = Y + ((long)dstIdx[e0] * 9 + n0) * 128;
  const _Float16* lbS = Wlds + (lane & 15) * MSGP;
  const _Float16* lbD = Wlds + (64 + (lane & 15)) * MSGP;

  v8f acc[4] = {};
#pragma unroll
  for (int k = 0; k < 128; k += 32) {
    v16h as = frag_a16(aS, k + ksA);
    v16h ad = frag_a16(aD, k + ksA);
#pragma unroll
    for (int t = 0; t < 4; ++t) {
      v16h fbs = frag_b16(lbS + t * 16 * MSGP, k + ksB);
      v16h fbd = frag_b16(lbD + t * 16 * MSGP, k + ksB);
      acc[t] = __builtin_amdgcn_wmma_f32_16x16x32_f16(false, as, false, fbs, (short)0, acc[t], false, false);
      acc[t] = __builtin_amdgcn_wmma_f32_16x16x32_f16(false, ad, false, fbd, (short)0, acc[t], false, false);
    }
  }

  if (!rowok) return;
  const int subr = (lane >> 4) << 3;
  const int cloc = lane & 15;
#pragma unroll
  for (int t = 0; t < 4; ++t) {
#pragma unroll
    for (int j = 0; j < 8; ++j) {
      const long row = row0 + subr + j;
      const int  col = col0 + t * 16 + cloc;
      const int  e   = (int)(row / 9);
      float v = acc[t][j] * (float)rad[(long)e * 128 + col];
      Mout[row * 128 + col] = (_Float16)v;
    }
  }
}

// ---------------------------------------------------------------------------
// Elementwise / scatter kernels
// ---------------------------------------------------------------------------
__global__ void zero_k(float* p, long n) {
  long t = (long)blockIdx.x * blockDim.x + threadIdx.x;
  if (t < n) p[t] = 0.0f;
}

__global__ void geom_k(const float* __restrict__ evec, float* __restrict__ dist,
                       float* __restrict__ sh, int E) {
  int e = blockIdx.x * blockDim.x + threadIdx.x;
  if (e >= E) return;
  float x = evec[3 * e], y = evec[3 * e + 1], z = evec[3 * e + 2];
  float d = sqrtf(x * x + y * y + z * z + 1e-12f);
  dist[e] = d;
  float ix = x / d, iy = y / d, iz = z / d;
  const float s3 = 1.7320508075688772f, s5 = 2.23606797749979f, s15 = 3.872983346207417f;
  float* o = sh + (long)e * 9;
  o[0] = 1.0f;       o[1] = s3 * ix;    o[2] = s3 * iy;    o[3] = s3 * iz;
  o[4] = s15 * ix * iy; o[5] = s15 * iy * iz;
  o[6] = 0.5f * s5 * (3.0f * iz * iz - 1.0f);
  o[7] = s15 * ix * iz;
  o[8] = 0.5f * s15 * (ix * ix - iy * iy);
}

__global__ void dex_k(const float* __restrict__ dist, _Float16* __restrict__ dex, int E) {
  long t = (long)blockIdx.x * blockDim.x + threadIdx.x;
  if (t >= (long)E * DPAD) return;
  int e = (int)(t / DPAD);
  int k = (int)(t - (long)e * DPAD);
  float v = 0.0f;
  if (k < DRBF) {
    const float step = 6.0f / 599.0f;
    const float gc = -0.5f / (2.0f * step * step);
    float dd = dist[e] - (float)k * step;
    v = __expf(gc * dd * dd);
  }
  dex[t] = (_Float16)v;
}

__global__ void convT_k(const float* __restrict__ W, _Float16* __restrict__ Wt,
                        int K, int N, int Kpad) {
  long t = (long)blockIdx.x * blockDim.x + threadIdx.x;
  if (t >= (long)N * Kpad) return;
  int n = (int)(t / Kpad);
  int k = (int)(t - (long)n * Kpad);
  Wt[t] = (_Float16)((k < K) ? W[(long)k * N + n] : 0.0f);
}

__global__ void scatter_deg_k(const _Float16* __restrict__ radDeg, const float* __restrict__ sh,
                              const int* __restrict__ dst, float* __restrict__ x, int E) {
  long t = (long)blockIdx.x * blockDim.x + threadIdx.x;
  if (t >= (long)E * 128) return;
  int e = (int)(t >> 7), c = (int)(t & 127);
  float r = (float)radDeg[t] * (1.0f / 12.0f);
  long base = (long)dst[e] * 9 * 128 + c;
  const float* she = sh + (long)e * 9;
#pragma unroll
  for (int n = 0; n < 9; ++n) atomicAdd(&x[base + n * 128], she[n] * r);
}

__global__ void addemb_k(const int* __restrict__ Z, const float* __restrict__ emb,
                         float* __restrict__ x, int Nn) {
  int t = blockIdx.x * blockDim.x + threadIdx.x;
  if (t >= Nn * 128) return;
  int i = t >> 7, c = t & 127;
  x[(long)i * 9 * 128 + c] += emb[(long)Z[i] * 128 + c];
}

__global__ __launch_bounds__(128) void rmsnorm_k(const float* __restrict__ x,
                                                 const float* __restrict__ gamma,
                                                 _Float16* __restrict__ y, int Nn) {
  __shared__ float red[128];
  int i = blockIdx.x, c = threadIdx.x;
  if (i >= Nn) return;
  const float* xp = x + (long)i * 9 * 128 + c;
  float vals[9];
  float s = 0.0f;
#pragma unroll
  for (int n = 0; n < 9; ++n) { float v = xp[n * 128]; vals[n] = v; s += v * v; }
  red[c] = s;
  __syncthreads();
  for (int off = 64; off > 0; off >>= 1) {
    if (c < off) red[c] += red[c + off];
    __syncthreads();
  }
  float g = gamma[c] * rsqrtf(red[0] / 1152.0f + 1e-6f);
  _Float16* yp = y + (long)i * 9 * 128 + c;
#pragma unroll
  for (int n = 0; n < 9; ++n) yp[n * 128] = (_Float16)(vals[n] * g);
}

__device__ __forceinline__ unsigned encf(float f) {
  unsigned u = __float_as_uint(f);
  return (u & 0x80000000u) ? ~u : (u | 0x80000000u);
}
__device__ __forceinline__ float decf(unsigned u) {
  return (u & 0x80000000u) ? __uint_as_float(u ^ 0x80000000u) : __uint_as_float(~u);
}

__global__ void logits_k(const _Float16* __restrict__ attn, const float* __restrict__ avec,
                         const int* __restrict__ dst, float* __restrict__ logits,
                         unsigned* __restrict__ nmax, int E) {
  int t = blockIdx.x * blockDim.x + threadIdx.x;
  if (t >= E * 8) return;
  int e = t >> 3, h = t & 7;
  const _Float16* ap = attn + (long)e * 256 + h * 32;
  float s = 0.0f;
#pragma unroll
  for (int a = 0; a < 32; ++a) s += (float)ap[a] * avec[h * 32 + a];
  s *= 0.17677669529663687f;  // 1/sqrt(32)
  logits[t] = s;
  atomicMax(&nmax[(long)dst[e] * 8 + h], encf(s));
}

__global__ void expden_k(const int* __restrict__ dst, float* __restrict__ logits,
                         const unsigned* __restrict__ nmax, float* __restrict__ den, int E) {
  int t = blockIdx.x * blockDim.x + threadIdx.x;
  if (t >= E * 8) return;
  int e = t >> 3, h = t & 7;
  float smax = decf(nmax[(long)dst[e] * 8 + h]);
  if (!(smax > -1e30f && smax < 1e30f)) smax = 0.0f;
  float ex = __expf(logits[t] - smax);
  logits[t] = ex;
  atomicAdd(&den[(long)dst[e] * 8 + h], ex);
}

__global__ void alpha_k(const int* __restrict__ dst, const float* __restrict__ ex,
                        const float* __restrict__ den, float* __restrict__ alpha, int E) {
  int t = blockIdx.x * blockDim.x + threadIdx.x;
  if (t >= E * 8) return;
  int e = t >> 3, h = t & 7;
  alpha[t] = ex[t] / (den[(long)dst[e] * 8 + h] + 1e-9f);
}

__global__ void gatefuse_k(_Float16* __restrict__ hbuf, const _Float16* __restrict__ gate, int Nn) {
  long t = (long)blockIdx.x * blockDim.x + threadIdx.x;
  if (t >= (long)Nn * 512) return;
  int i = (int)(t >> 9), f = (int)(t & 511);
  float g = (float)gate[(long)i * 512 + f];
  _Float16* hp = hbuf + (long)i * 9 * 512 + f;
  float h0 = (float)hp[0];
  hp[0] = (_Float16)siluf(h0);
#pragma unroll
  for (int n = 1; n < 9; ++n) hp[(long)n * 512] = (_Float16)((float)hp[(long)n * 512] * g);
}

__global__ __launch_bounds__(256) void node_e_k(const _Float16* __restrict__ hbuf,
                                                const float* __restrict__ w,
                                                float* __restrict__ node_e, int Nn) {
  int wave = threadIdx.x >> 5, lane = threadIdx.x & 31;
  int i = blockIdx.x * 8 + wave;
  if (i >= Nn) return;
  const _Float16* hp = hbuf + (long)i * 9 * 512;
  float s = 0.0f;
  for (int f = lane; f < 512; f += 32) s += (float)hp[f] * w[f];
  for (int off = 16; off > 0; off >>= 1) s += __shfl_down(s, off, 32);
  if (lane == 0) node_e[i] = s;
}

__global__ void graph_k(const int* __restrict__ batch, const float* __restrict__ node_e,
                        float* __restrict__ energy, float* __restrict__ cnt, int Nn) {
  int i = blockIdx.x * blockDim.x + threadIdx.x;
  if (i >= Nn) return;
  atomicAdd(&energy[batch[i]], node_e[i]);
  atomicAdd(&cnt[batch[i]], 1.0f);
}

__global__ void out_k(const float* __restrict__ energy, const float* __restrict__ cnt,
                      float* __restrict__ out) {
  int g = threadIdx.x;
  if (g < NGRAPH) out[g] = energy[g] / fmaxf(cnt[g], 1.0f);
}

// ---------------------------------------------------------------------------
// Host launcher
// ---------------------------------------------------------------------------
extern "C" void kernel_launch(void* const* d_in, const int* in_sizes, int n_in,
                              void* d_out, int out_size, void* d_ws, size_t ws_size,
                              hipStream_t stream) {
  (void)in_sizes; (void)n_in; (void)out_size;
  const int E = NEDGES, Nn = NATOMS;

  const int*   Zn     = (const int*)d_in[0];
  const int*   srcIdx = (const int*)d_in[1];
  const int*   dstIdx = srcIdx + E;
  const int*   batch  = (const int*)d_in[2];
  const float* evec   = (const float*)d_in[3];
  const float* aemb   = (const float*)d_in[4];
  const float* deg_w1 = (const float*)d_in[5];
  const float* deg_b1 = (const float*)d_in[6];
  const float* deg_w2 = (const float*)d_in[7];
  const float* deg_b2 = (const float*)d_in[8];
  const float* rad_w1 = (const float*)d_in[9];
  const float* rad_b1 = (const float*)d_in[10];
  const float* rad_w2 = (const float*)d_in[11];
  const float* rad_b2 = (const float*)d_in[12];
  const float* w_src  = (const float*)d_in[13];
  const float* w_dst  = (const float*)d_in[14];
  const float* alpha_w = (const float*)d_in[15];
  const float* alpha_v = (const float*)d_in[16];
  const float* value_w = (const float*)d_in[17];
  const float* proj_w  = (const float*)d_in[18];
  const float* g_attn  = (const float*)d_in[19];
  const float* g_ffn   = (const float*)d_in[20];
  const float* ffn_w1  = (const float*)d_in[21];
  const float* ffn_gw  = (const float*)d_in[22];
  const float* ffn_w2  = (const float*)d_in[23];
  const float* g_final = (const float*)d_in[24];
  const float* e_w1    = (const float*)d_in[25];
  const float* e_gw    = (const float*)d_in[26];
  const float* e_w2    = (const float*)d_in[27];
  float* out = (float*)d_out;

  // ---- workspace bump allocator ----
  char* wp = (char*)d_ws;
  auto alloc = [&](size_t bytes) -> void* {
    void* p = (void*)wp;
    wp += (bytes + 255) & ~(size_t)255;
    return p;
  };
  float*     dist   = (float*)alloc((size_t)E * 4);
  float*     sh     = (float*)alloc((size_t)E * 9 * 4);
  _Float16*  dexh   = (_Float16*)alloc((size_t)E * DPAD * 2);
  float*     x      = (float*)alloc((size_t)Nn * 9 * 128 * 4);
  _Float16*  y      = (_Float16*)alloc((size_t)Nn * 9 * 128 * 2);
  _Float16*  tmp1   = (_Float16*)alloc((size_t)E * 128 * 2);
  _Float16*  rad    = (_Float16*)alloc((size_t)E * 128 * 2);
  _Float16*  m      = (_Float16*)alloc((size_t)E * 9 * 128 * 2);
  _Float16*  attn   = (_Float16*)alloc((size_t)E * 256 * 2);
  float*     logits = (float*)alloc((size_t)E * 8 * 4);
  float*     alphaB = (float*)alloc((size_t)E * 8 * 4);
  unsigned*  nmax   = (unsigned*)alloc((size_t)Nn * 8 * 4);
  float*     den    = (float*)alloc((size_t)Nn * 8 * 4);
  float*     agg    = (float*)alloc((size_t)Nn * 9 * 128 * 4);
  _Float16*  hbuf   = (_Float16*)alloc((size_t)Nn * 9 * 512 * 2);
  _Float16*  gate   = (_Float16*)alloc((size_t)Nn * 512 * 2);
  float*     node_e = (float*)alloc((size_t)Nn * 4);
  float*     energy = (float*)alloc(128 * 4);
  float*     cnt    = energy + 64;
  // f16 transposed weights
  _Float16* degw1T = (_Float16*)alloc((size_t)128 * DPAD * 2);
  _Float16* degw2T = (_Float16*)alloc((size_t)128 * 128 * 2);
  _Float16* radw1T = (_Float16*)alloc((size_t)NLAY * 128 * DPAD * 2);
  _Float16* radw2T = (_Float16*)alloc((size_t)NLAY * 128 * 128 * 2);
  _Float16* wsrcT  = (_Float16*)alloc((size_t)NLAY * 128 * 128 * 2);
  _Float16* wdstT  = (_Float16*)alloc((size_t)NLAY * 128 * 128 * 2);
  _Float16* alphaT = (_Float16*)alloc((size_t)NLAY * 256 * 128 * 2);
  _Float16* valueT = (_Float16*)alloc((size_t)NLAY * 128 * 128 * 2);
  _Float16* projT  = (_Float16*)alloc((size_t)NLAY * 128 * 128 * 2);
  _Float16* fw1T   = (_Float16*)alloc((size_t)NLAY * 512 * 128 * 2);
  _Float16* fgT    = (_Float16*)alloc((size_t)NLAY * 512 * 128 * 2);
  _Float16* fw2T   = (_Float16*)alloc((size_t)NLAY * 128 * 512 * 2);
  _Float16* ew1T   = (_Float16*)alloc((size_t)512 * 128 * 2);
  _Float16* egT    = (_Float16*)alloc((size_t)512 * 128 * 2);
  if ((size_t)(wp - (char*)d_ws) > ws_size) return;  // workspace too small

  auto cdiv = [](long a, long b) { return (int)((a + b - 1) / b); };

#define GEMM(AF32, EPI, Aptr, RS, Bptr, K, M, NCOLS, BIASP, ALP, DSTP, AGGP, COUTP, LDC, ADDP)     \
  gemm_k<AF32, EPI><<<dim3(cdiv((M), 128), (NCOLS) / 64), 256, GEMM_LDS_BYTES, stream>>>(           \
      (const void*)(Aptr), (long)(RS), (Bptr), (K), (M), (BIASP), (ALP), (DSTP), (AGGP),            \
      (void*)(COUTP), (long)(LDC), (ADDP))

  auto convT = [&](const float* s, _Float16* d, int K, int N, int Kpad) {
    long tot = (long)N * Kpad;
    convT_k<<<cdiv(tot, 256), 256, 0, stream>>>(s, d, K, N, Kpad);
  };

  // ---- weight conversion (f32 -> f16, transposed, K padded) ----
  convT(deg_w1, degw1T, DRBF, 128, DPAD);
  convT(deg_w2, degw2T, 128, 128, 128);
  for (int l = 0; l < NLAY; ++l) {
    convT(rad_w1 + (long)l * DRBF * 128, radw1T + (long)l * 128 * DPAD, DRBF, 128, DPAD);
    convT(rad_w2 + (long)l * 128 * 128, radw2T + (long)l * 128 * 128, 128, 128, 128);
    convT(w_src + (long)l * 128 * 128, wsrcT + (long)l * 128 * 128, 128, 128, 128);
    convT(w_dst + (long)l * 128 * 128, wdstT + (long)l * 128 * 128, 128, 128, 128);
    convT(alpha_w + (long)l * 128 * 256, alphaT + (long)l * 256 * 128, 128, 256, 128);
    convT(value_w + (long)l * 128 * 128, valueT + (long)l * 128 * 128, 128, 128, 128);
    convT(proj_w + (long)l * 128 * 128, projT + (long)l * 128 * 128, 128, 128, 128);
    convT(ffn_w1 + (long)l * 128 * 512, fw1T + (long)l * 512 * 128, 128, 512, 128);
    convT(ffn_gw + (long)l * 128 * 512, fgT + (long)l * 512 * 128, 128, 512, 128);
    convT(ffn_w2 + (long)l * 512 * 128, fw2T + (long)l * 128 * 512, 512, 128, 512);
  }
  convT(e_w1, ew1T, 128, 512, 128);
  convT(e_gw, egT, 128, 512, 128);

  // ---- geometry, smearing ----
  geom_k<<<cdiv(E, 256), 256, 0, stream>>>(evec, dist, sh, E);
  dex_k<<<cdiv((long)E * DPAD, 256), 256, 0, stream>>>(dist, dexh, E);

  // ---- degree embedding: rad_deg = silu(dex@w1+b1)@w2+b2 ; x = scatter/12 + emb ----
  GEMM(false, 1, dexh, DPAD, degw1T, DPAD, E, 128, deg_b1, nullptr, nullptr, nullptr, tmp1, 128, nullptr);
  GEMM(false, 2, tmp1, 128, degw2T, 128, E, 128, deg_b2, nullptr, nullptr, nullptr, rad, 128, nullptr);
  zero_k<<<cdiv((long)Nn * 9 * 128, 256), 256, 0, stream>>>(x, (long)Nn * 9 * 128);
  scatter_deg_k<<<cdiv((long)E * 128, 256), 256, 0, stream>>>(rad, sh, dstIdx, x, E);
  addemb_k<<<cdiv(Nn * 128, 256), 256, 0, stream>>>(Zn, aemb, x, Nn);

  // ---- layers ----
  for (int l = 0; l < NLAY; ++l) {
    zero_k<<<cdiv((long)Nn * 8, 256), 256, 0, stream>>>((float*)nmax, (long)Nn * 8);
    zero_k<<<cdiv((long)Nn * 8, 256), 256, 0, stream>>>(den, (long)Nn * 8);
    zero_k<<<cdiv((long)Nn * 9 * 128, 256), 256, 0, stream>>>(agg, (long)Nn * 9 * 128);

    rmsnorm_k<<<Nn, 128, 0, stream>>>(x, g_attn + l * 128, y, Nn);

    // radial MLP
    GEMM(false, 1, dexh, DPAD, radw1T + (long)l * 128 * DPAD, DPAD, E, 128,
         rad_b1 + l * 128, nullptr, nullptr, nullptr, tmp1, 128, nullptr);
    GEMM(false, 2, tmp1, 128, radw2T + (long)l * 128 * 128, 128, E, 128,
         rad_b2 + l * 128, nullptr, nullptr, nullptr, rad, 128, nullptr);

    // messages m = (y[src]@Wsrc + y[dst]@Wdst) * rad
    gemm_msg_k<<<dim3(cdiv((long)E * 9, 128), 2), 256, MSG_LDS_BYTES, stream>>>(
        y, srcIdx, dstIdx, wsrcT + (long)l * 128 * 128, wdstT + (long)l * 128 * 128,
        rad, m, E * 9);

    // attention logits
    GEMM(false, 3, m, 1152, alphaT + (long)l * 256 * 128, 128, E, 256,
         nullptr, nullptr, nullptr, nullptr, attn, 256, nullptr);
    logits_k<<<cdiv(E * 8, 256), 256, 0, stream>>>(attn, alpha_v + l * 256, dstIdx, logits, nmax, E);
    expden_k<<<cdiv(E * 8, 256), 256, 0, stream>>>(dstIdx, logits, nmax, den, E);
    alpha_k<<<cdiv(E * 8, 256), 256, 0, stream>>>(dstIdx, logits, den, alphaB, E);

    // values: v = m@Wv, scaled by alpha, scattered into agg
    GEMM(false, 5, m, 128, valueT + (long)l * 128 * 128, 128, E * 9, 128,
         nullptr, alphaB, dstIdx, agg, nullptr, 128, nullptr);

    // x += agg @ Wproj
    GEMM(true, 4, agg, 128, projT + (long)l * 128 * 128, 128, Nn * 9, 128,
         nullptr, nullptr, nullptr, nullptr, nullptr, 128, x);

    // FFN (SwiGLU across NC with gate from n=0)
    rmsnorm_k<<<Nn, 128, 0, stream>>>(x, g_ffn + l * 128, y, Nn);
    GEMM(false, 0, y, 128, fw1T + (long)l * 512 * 128, 128, Nn * 9, 512,
         nullptr, nullptr, nullptr, nullptr, hbuf, 512, nullptr);
    GEMM(false, 3, y, 1152, fgT + (long)l * 512 * 128, 128, Nn, 512,
         nullptr, nullptr, nullptr, nullptr, gate, 512, nullptr);
    gatefuse_k<<<cdiv((long)Nn * 512, 256), 256, 0, stream>>>(hbuf, gate, Nn);
    GEMM(false, 4, hbuf, 512, fw2T + (long)l * 128 * 512, 512, Nn * 9, 128,
         nullptr, nullptr, nullptr, nullptr, nullptr, 128, x);
  }

  // ---- energy head ----
  rmsnorm_k<<<Nn, 128, 0, stream>>>(x, g_final, y, Nn);
  GEMM(false, 0, y, 128, ew1T, 128, Nn * 9, 512,
       nullptr, nullptr, nullptr, nullptr, hbuf, 512, nullptr);
  GEMM(false, 3, y, 1152, egT, 128, Nn, 512,
       nullptr, nullptr, nullptr, nullptr, gate, 512, nullptr);
  gatefuse_k<<<cdiv((long)Nn * 512, 256), 256, 0, stream>>>(hbuf, gate, Nn);
  node_e_k<<<cdiv(Nn, 8), 256, 0, stream>>>(hbuf, e_w2, node_e, Nn);

  zero_k<<<1, 128, 0, stream>>>(energy, 128);
  graph_k<<<cdiv(Nn, 256), 256, 0, stream>>>(batch, node_e, energy, cnt, Nn);
  out_k<<<1, 64, 0, stream>>>(energy, cnt, out);
#undef GEMM
}
